// SpatialConvolutionLayer_28673201668173
// MI455X (gfx1250) — matrix operand order repaired and checked by reading, not compiled, not run on hardware
//
#include <hip/hip_runtime.h>
#include <math.h>

// Shapes from the reference: B=4, T=16, N=1024, F=64  -> S = B*T = 64 slabs.
#define SLABS 64
#define NN    1024
#define FF    64
#define EPSF  1e-8f

typedef __attribute__((ext_vector_type(16))) _Float16 v16h;
typedef __attribute__((ext_vector_type(8)))  float    v8f;

__device__ __forceinline__ v8f wmma_f16(v16h a, v16h b, v8f c) {
    // v_wmma_f32_16x16x32_f16: D = A(16x32 f16) * B(32x16 f16) + C(16x16 f32)
    return __builtin_amdgcn_wmma_f32_16x16x32_f16(false, a, false, b, (short)0, c, false, false);
}

// ---------------------------------------------------------------------------
// Kernel 1: dinv[row] = rsqrt( sum_m |A[row,m]| + 1 )   (row of A_hat = |A|+I)
// One wave per row; coalesced float4 loads; shuffle reduction.
// ---------------------------------------------------------------------------
__global__ void dinv_kernel(const float* __restrict__ A, float* __restrict__ dinv) {
    const int wid  = threadIdx.x >> 5;
    const int lane = threadIdx.x & 31;
    const long row = (long)blockIdx.x * 8 + wid;          // 64*1024 rows total
    const float4* Ar = (const float4*)(A + row * (long)NN);
    float s = 0.f;
#pragma unroll
    for (int k = 0; k < 8; ++k) {                          // 1024/4/32 = 8
        float4 v = Ar[lane + 32 * k];
        s += fabsf(v.x) + fabsf(v.y) + fabsf(v.z) + fabsf(v.w);
    }
#pragma unroll
    for (int m = 1; m <= 16; m <<= 1) s += __shfl_xor(s, m, 32);
    if (lane == 0) dinv[row] = rsqrtf(s + 1.0f);
}

// ---------------------------------------------------------------------------
// Kernel 2: Q/K/V = X @ W^T + b (WMMA f16), stored as f16; row norms of Q,K.
// Block = 128 threads (4 waves); each wave owns a 16-row tile.
// W is pre-converted to f16 in LDS (kills register spills); output tiles are
// staged in LDS and stored with coalesced 16-B writes.
// ---------------------------------------------------------------------------
__global__ void qkv_kernel(const float* __restrict__ X,
                           const float* __restrict__ Wq, const float* __restrict__ bq,
                           const float* __restrict__ Wk, const float* __restrict__ bk,
                           const float* __restrict__ Wv, const float* __restrict__ bv,
                           _Float16* __restrict__ Qh, _Float16* __restrict__ Kh,
                           _Float16* __restrict__ Vh,
                           float* __restrict__ qn, float* __restrict__ kn) {
    __shared__ __align__(32) _Float16 WhS[3 * FF * FF];   // 24 KB: Wq,Wk,Wv in f16
    __shared__ float bS[3 * FF];
    __shared__ __align__(16) _Float16 OtS[4][16 * FF];    // 8 KB: per-wave out tile

    const int slab  = blockIdx.x;
    const int wid   = threadIdx.x >> 5;
    const int lane  = threadIdx.x & 31;
    const int ln    = lane & 15;
    const int hi    = lane >> 4;          // 0: low half, 1: high half of wave
    const int mbase = blockIdx.y * 64 + wid * 16;

    // Cooperative W -> f16 LDS conversion (block-wide, coalesced).
    const float* Wsrc[3] = {Wq, Wk, Wv};
#pragma unroll
    for (int mat = 0; mat < 3; ++mat)
        for (int i = threadIdx.x; i < FF * FF; i += 128)
            WhS[mat * FF * FF + i] = (_Float16)Wsrc[mat][i];
    if (threadIdx.x < FF) {
        bS[threadIdx.x]          = bq[threadIdx.x];
        bS[FF + threadIdx.x]     = bk[threadIdx.x];
        bS[2 * FF + threadIdx.x] = bv[threadIdx.x];
    }
    __syncthreads();

    const float* Xs = X + (long)slab * NN * FF;

    // A-fragments of X (16x32 f16 each): lane holds row m=ln; K chunk by half.
    v16h a0, a1;
    const int kb = hi * 16;
#pragma unroll
    for (int e = 0; e < 8; ++e) {
        a0[e]     = (_Float16)Xs[(long)(mbase + ln) * FF + kb + e];
        a0[e + 8] = (_Float16)Xs[(long)(mbase + ln) * FF + 8 + kb + e];
        a1[e]     = (_Float16)Xs[(long)(mbase + ln) * FF + 32 + kb + e];
        a1[e + 8] = (_Float16)Xs[(long)(mbase + ln) * FF + 40 + kb + e];
    }

    _Float16* Os[3] = {Qh, Kh, Vh};
    float*    Ns[2] = {qn, kn};
    const int kb2 = hi * 16;

#pragma unroll
    for (int mat = 0; mat < 3; ++mat) {
        float nrm[8];
#pragma unroll
        for (int r = 0; r < 8; ++r) nrm[r] = 0.f;
#pragma unroll
        for (int nt = 0; nt < 4; ++nt) {
            // B fragments of W^T (32x16): lane holds col n=nt*16+ln; K by half.
            const int n = nt * 16 + ln;
            v16h b0 = *(const v16h*)(WhS + mat * FF * FF + n * FF + kb2);
            v16h b1 = *(const v16h*)(WhS + mat * FF * FF + n * FF + 32 + kb2);
            v8f c = {0.f, 0.f, 0.f, 0.f, 0.f, 0.f, 0.f, 0.f};
            c = wmma_f16(a0, b0, c);
            c = wmma_f16(a1, b1, c);
            const float bias = bS[mat * FF + n];
#pragma unroll
            for (int r = 0; r < 8; ++r) {
                float v = c[r] + bias;
                if (mat < 2) nrm[r] += v * v;
                OtS[wid][(r + hi * 8) * FF + n] = (_Float16)v;  // C layout
            }
        }
        asm volatile("s_wait_dscnt 0" ::: "memory");  // tile writes -> reads
        // Coalesced store: tile = 16 consecutive full rows = contiguous 2 KB.
        {
            uint4* dst = (uint4*)(Os[mat] + ((long)slab * NN + mbase) * FF);
            const uint4* src = (const uint4*)OtS[wid];
#pragma unroll
            for (int it = 0; it < 4; ++it)
                dst[it * 32 + lane] = src[it * 32 + lane];
        }
        if (mat < 2) {
#pragma unroll
            for (int r = 0; r < 8; ++r) {
                float v = nrm[r];
#pragma unroll
                for (int m = 1; m <= 8; m <<= 1) v += __shfl_xor(v, m, 32);
                if (ln == 0)
                    Ns[mat][(long)slab * NN + mbase + r + hi * 8] = sqrtf(v);
            }
        }
    }
}

// ---------------------------------------------------------------------------
// Kernel 3: flash attention with A_norm * cosine-similarity logits.
// Block = 256 threads (8 waves); wave owns 16 rows; block loops 32-col tiles.
// ---------------------------------------------------------------------------
__global__ void attn_kernel(const float* __restrict__ A,
                            const _Float16* __restrict__ Qh,
                            const _Float16* __restrict__ Kh,
                            const _Float16* __restrict__ Vh,
                            const float* __restrict__ qn, const float* __restrict__ kn,
                            const float* __restrict__ dinv,
                            float* __restrict__ out) {
    __shared__ __align__(32) _Float16 Kt[32 * 64];        // K tile (rows jb..jb+31)
    __shared__ __align__(32) _Float16 Vt[32 * 64];        // V tile
    __shared__ __align__(32) _Float16 Pt[8][16 * 32];     // per-wave P transpose buf

    const int slab  = blockIdx.x;
    const int wid   = threadIdx.x >> 5;
    const int lane  = threadIdx.x & 31;
    const int ln    = lane & 15;
    const int hi    = lane >> 4;
    const int mbase = blockIdx.y * 128 + wid * 16;

    const float*    As = A  + (long)slab * NN * NN;
    const _Float16* Qs = Qh + (long)slab * NN * FF;
    const _Float16* Ks = Kh + (long)slab * NN * FF;
    const _Float16* Vs = Vh + (long)slab * NN * FF;

    // Q A-fragments (persistent): row m = ln, F split 0..31 / 32..63.
    v16h qa0, qa1;
    const int kb = hi * 16;
#pragma unroll
    for (int e = 0; e < 8; ++e) {
        qa0[e]     = Qs[(long)(mbase + ln) * FF + kb + e];
        qa0[e + 8] = Qs[(long)(mbase + ln) * FF + 8 + kb + e];
        qa1[e]     = Qs[(long)(mbase + ln) * FF + 32 + kb + e];
        qa1[e + 8] = Qs[(long)(mbase + ln) * FF + 40 + kb + e];
    }

    float qd[8], dm[8], rmax[8], rsum[8];
#pragma unroll
    for (int r = 0; r < 8; ++r) {
        const int m = mbase + r + hi * 8;
        qd[r]   = qn[(long)slab * NN + m];
        dm[r]   = dinv[(long)slab * NN + m];
        rmax[r] = -INFINITY;
        rsum[r] = 0.f;
    }
    v8f acc[4];
#pragma unroll
    for (int nt = 0; nt < 4; ++nt) acc[nt] = (v8f){0,0,0,0,0,0,0,0};

    for (int jt = 0; jt < 32; ++jt) {
        const int jb = jt * 32;
        // Stage K/V tiles: 32 rows x 64 halves = 256 x 16B, one uint4/thread.
        ((uint4*)Kt)[threadIdx.x] = ((const uint4*)(Ks + (long)jb * FF))[threadIdx.x];
        ((uint4*)Vt)[threadIdx.x] = ((const uint4*)(Vs + (long)jb * FF))[threadIdx.x];
        __syncthreads();

        if (jt + 1 < 32)   // speculative prefetch of next A columns for our rows
            __builtin_prefetch(As + (long)(mbase + ln) * NN + jb + 32, 0, 0);

        float l0[8], l1[8];
#pragma unroll
        for (int ch = 0; ch < 2; ++ch) {   // two 16-column halves of the tile
            const int cl = ch * 16 + ln;                       // col within tile
            // K^T B-fragments from LDS (contiguous 32B per lane).
            v16h b0 = *(const v16h*)(Kt + cl * 64 + hi * 16);
            v16h b1 = *(const v16h*)(Kt + cl * 64 + 32 + hi * 16);
            v8f s = {0.f, 0.f, 0.f, 0.f, 0.f, 0.f, 0.f, 0.f};
            s = wmma_f16(qa0, b0, s);
            s = wmma_f16(qa1, b1, s);

            const int   col = jb + cl;
            const float knc = kn[(long)slab * NN + col];
            const float dc  = dinv[(long)slab * NN + col];
            float* lv = ch ? l1 : l0;
#pragma unroll
            for (int r = 0; r < 8; ++r) {
                const int  m   = mbase + r + hi * 8;
                float aval = fabsf(As[(long)m * NN + col]) + (m == col ? 1.f : 0.f);
                float cosv = s[r] / fmaxf(qd[r] * knc, EPSF);
                lv[r] = dm[r] * dc * aval * cosv;
            }
        }

        // Online softmax update + write P (f16) to per-wave LDS buffer.
#pragma unroll
        for (int r = 0; r < 8; ++r) {
            float tm = fmaxf(l0[r], l1[r]);
#pragma unroll
            for (int m = 1; m <= 8; m <<= 1) tm = fmaxf(tm, __shfl_xor(tm, m, 32));
            const float nmax  = fmaxf(rmax[r], tm);
            const float scale = __expf(rmax[r] - nmax);
            const float p0 = __expf(l0[r] - nmax);
            const float p1 = __expf(l1[r] - nmax);
            float ts = p0 + p1;
#pragma unroll
            for (int m = 1; m <= 8; m <<= 1) ts += __shfl_xor(ts, m, 32);
            rsum[r] = rsum[r] * scale + ts;
            rmax[r] = nmax;
#pragma unroll
            for (int nt = 0; nt < 4; ++nt) acc[nt][r] *= scale;
            const int m = r + hi * 8;
            Pt[wid][m * 32 + ln]      = (_Float16)p0;
            Pt[wid][m * 32 + 16 + ln] = (_Float16)p1;
        }
        asm volatile("s_wait_dscnt 0" ::: "memory");  // P stores visible to reads

        // P as A-fragment (16x32), C-layout -> A-layout via LDS.
        v16h pa;
#pragma unroll
        for (int e = 0; e < 8; ++e) {
            pa[e]     = Pt[wid][ln * 32 + kb + e];
            pa[e + 8] = Pt[wid][ln * 32 + 8 + kb + e];
        }
        // acc += P * V  (4 column tiles of F)
#pragma unroll
        for (int nt = 0; nt < 4; ++nt) {
            v16h bv;
#pragma unroll
            for (int e = 0; e < 16; ++e)
                bv[e] = Vt[(hi * 16 + e) * 64 + nt * 16 + ln];
            acc[nt] = wmma_f16(pa, bv, acc[nt]);
        }
        __syncthreads();
    }

    // Normalize, ReLU, store.
#pragma unroll
    for (int nt = 0; nt < 4; ++nt) {
#pragma unroll
        for (int r = 0; r < 8; ++r) {
            const int m = mbase + r + hi * 8;
            float v = acc[nt][r] / rsum[r];
            out[((long)slab * NN + m) * FF + nt * 16 + ln] = fmaxf(v, 0.f);
        }
    }
}

// ---------------------------------------------------------------------------
extern "C" void kernel_launch(void* const* d_in, const int* in_sizes, int n_in,
                              void* d_out, int out_size, void* d_ws, size_t ws_size,
                              hipStream_t stream) {
    const float* X  = (const float*)d_in[0];   // hidden_x [4,16,1024,64]
    const float* A  = (const float*)d_in[1];   // A        [4,16,1024,1024]
    const float* Wq = (const float*)d_in[2];
    const float* bq = (const float*)d_in[3];
    const float* Wk = (const float*)d_in[4];
    const float* bk = (const float*)d_in[5];
    const float* Wv = (const float*)d_in[6];
    const float* bv = (const float*)d_in[7];
    float* out = (float*)d_out;

    // Workspace layout (needs ~24.8 MB): Qh/Kh/Vh f16, qn/kn/dinv f32.
    char* w = (char*)d_ws;
    _Float16* Qh = (_Float16*)w; w += (size_t)SLABS * NN * FF * sizeof(_Float16);
    _Float16* Kh = (_Float16*)w; w += (size_t)SLABS * NN * FF * sizeof(_Float16);
    _Float16* Vh = (_Float16*)w; w += (size_t)SLABS * NN * FF * sizeof(_Float16);
    float* qn   = (float*)w; w += (size_t)SLABS * NN * sizeof(float);
    float* kn   = (float*)w; w += (size_t)SLABS * NN * sizeof(float);
    float* dinv = (float*)w; w += (size_t)SLABS * NN * sizeof(float);

    dinv_kernel<<<dim3(SLABS * NN / 8), dim3(256), 0, stream>>>(A, dinv);
    qkv_kernel<<<dim3(SLABS, NN / 64), dim3(128), 0, stream>>>(
        X, Wq, bq, Wk, bk, Wv, bv, Qh, Kh, Vh, qn, kn);
    attn_kernel<<<dim3(SLABS, NN / 128), dim3(256), 0, stream>>>(
        A, Qh, Kh, Vh, qn, kn, dinv, out);
}
